// TransformerEncoderLayer_49074296324470
// MI455X (gfx1250) — matrix operand order repaired and checked by reading, not compiled, not run on hardware
//
#include <hip/hip_runtime.h>
#include <hip/hip_bf16.h>

#define D_MODEL 1024
#define SEQ     1024
#define BATCH   8
#define NHEAD   16
#define HD      64
#define NEXP    4
#define DFF_E   1024
#define NTOK    (BATCH * SEQ)

typedef __attribute__((ext_vector_type(16))) __bf16 v16bf;
typedef __attribute__((ext_vector_type(8)))  float  v8f;

union Frag {
    v16bf v;
    unsigned int   u[8];
    unsigned short s[16];
};

__device__ __forceinline__ unsigned short f2bf(float f) {
    unsigned int u = __float_as_uint(f);
    unsigned int r = u + 0x7FFFu + ((u >> 16) & 1u);   // round-to-nearest-even
    return (unsigned short)(r >> 16);
}

// ---------------------------------------------------------------- converts
__global__ __launch_bounds__(256) void cvt_bf16(const float* __restrict__ in,
                                                unsigned short* __restrict__ out, int n) {
    int i = blockIdx.x * blockDim.x + threadIdx.x;
    if (i < n) out[i] = f2bf(in[i]);
}

__global__ __launch_bounds__(256) void prep_kernel(const float* __restrict__ src,
                                                   const float* __restrict__ pos,
                                                   unsigned short* __restrict__ qkb,
                                                   unsigned short* __restrict__ srcb, int n) {
    int i = blockIdx.x * blockDim.x + threadIdx.x;
    if (i < n) {
        float s = src[i];
        qkb[i]  = f2bf(s + pos[i]);
        srcb[i] = f2bf(s);
    }
}

// ---------------------------------------------------------------- GEMM
// C[M,N] = A[M,K](bf16) x W[N,K](bf16)^T + bias[N]
// Register-blocked 32x64 per wave (2 A-frags x 4 B-frags = 8 WMMAs / K-step).
// Block = 8 waves (2 M x 4 N) -> 64x256 tile. The 64x32 A tile is staged in LDS
// via async global->LDS b128 copies (double-buffered, ASYNCcnt-tracked) so the
// 4 waves sharing an M strip read A from LDS instead of re-fetching from L2.
// EP: 0 = store f32 | 1 = store bf16 | 2 = relu -> bf16 | 3 = store f32 iff chosen[row]==expert
template <int EP>
__global__ __launch_bounds__(256) void gemm_bf16_nt(const unsigned short* __restrict__ A,
                                                    const unsigned short* __restrict__ W,
                                                    const float* __restrict__ bias,
                                                    void* __restrict__ out,
                                                    int M, int N, int K,
                                                    const int* __restrict__ chosen, int expert) {
    const int tid  = threadIdx.x;
    const int lane = tid & 31;
    const int wave = tid >> 5;
    const int tm = blockIdx.y * 64 + (wave >> 2) * 32;   // 32 rows per wave
    const int tn = blockIdx.x * 256 + (wave & 3) * 64;   // 64 cols per wave

    const int l15 = lane & 15;
    const int akb = (lane < 16) ? 0 : 8;     // A K-base per lane half
    const int bkb = (lane < 16) ? 0 : 8;     // B K-base per lane half

    const unsigned short* brow[4];
#pragma unroll
    for (int t = 0; t < 4; ++t) brow[t] = W + (size_t)(tn + t * 16 + l15) * K;

    // ---- LDS staging of the 64-row A tile (double buffered, 2 x 4KB) ----
    __shared__ unsigned short tileA[2][64 * 32];
    const unsigned short* Ablk = A + (size_t)(blockIdx.y * 64) * K;
    const int cr = tid >> 2;          // copy row   0..63
    const int cc = (tid & 3) * 8;     // copy halves 0,8,16,24  (16B each)

    auto issue_copy = [&](int buf, int k0) {
        unsigned lds_addr = (unsigned)(size_t)(void*)&tileA[buf][cr * 32 + cc];
        unsigned long long ga =
            (unsigned long long)(size_t)(Ablk + (size_t)cr * K + k0 + cc);
        asm volatile("global_load_async_to_lds_b128 %0, %1, off"
                     :: "v"(lds_addr), "v"(ga) : "memory");
    };

    v8f acc[2][4];
#pragma unroll
    for (int mi = 0; mi < 2; ++mi)
#pragma unroll
        for (int t = 0; t < 4; ++t)
            acc[mi][t] = (v8f){0.f, 0.f, 0.f, 0.f, 0.f, 0.f, 0.f, 0.f};

    issue_copy(0, 0);

    const int KT = K >> 5;
    for (int kt = 0; kt < KT; ++kt) {
        asm volatile("s_wait_asynccnt 0" ::: "memory");
        __syncthreads();
        if (kt + 1 < KT) issue_copy((kt + 1) & 1, (kt + 1) * 32);

        const int k0 = kt * 32;
        const unsigned short* la0 = &tileA[kt & 1][((wave >> 2) * 32 + l15) * 32];
        const unsigned short* la1 = la0 + 16 * 32;

        Frag a0, a1, b[4];
#pragma unroll
        for (int j = 0; j < 4; ++j) {
            a0.u[j]     = *(const unsigned int*)(la0 + akb + 2 * j);
            a0.u[j + 4] = *(const unsigned int*)(la0 + akb + 16 + 2 * j);
            a1.u[j]     = *(const unsigned int*)(la1 + akb + 2 * j);
            a1.u[j + 4] = *(const unsigned int*)(la1 + akb + 16 + 2 * j);
        }
#pragma unroll
        for (int t = 0; t < 4; ++t)
#pragma unroll
            for (int j = 0; j < 8; ++j)
                b[t].u[j] = *(const unsigned int*)(brow[t] + k0 + bkb + 2 * j + ((j >> 2) << 3));

#pragma unroll
        for (int t = 0; t < 4; ++t) {
            acc[0][t] = __builtin_amdgcn_wmma_f32_16x16x32_bf16(false, a0.v, false, b[t].v,
                                                                (short)0, acc[0][t], false, false);
            acc[1][t] = __builtin_amdgcn_wmma_f32_16x16x32_bf16(false, a1.v, false, b[t].v,
                                                                (short)0, acc[1][t], false, false);
        }
    }

#pragma unroll
    for (int t = 0; t < 4; ++t) {
        const int   nidx = tn + t * 16 + l15;
        const float bn   = bias[nidx];
#pragma unroll
        for (int mi = 0; mi < 2; ++mi) {
            const int mbase = tm + mi * 16 + ((lane < 16) ? 0 : 8);
#pragma unroll
            for (int j = 0; j < 8; ++j) {
                float  val = acc[mi][t][j] + bn;
                size_t idx = (size_t)(mbase + j) * N + nidx;
                if (EP == 0) {
                    ((float*)out)[idx] = val;
                } else if (EP == 1) {
                    ((unsigned short*)out)[idx] = f2bf(val);
                } else if (EP == 2) {
                    ((unsigned short*)out)[idx] = f2bf(fmaxf(val, 0.f));
                } else {  // EP == 3
                    if (chosen[mbase + j] == expert) ((float*)out)[idx] = val;
                }
            }
        }
    }
}

// ---------------------------------------------------------------- flash attention
// One wave handles 16 query rows of one (b,h); streams keys in chunks of 32.
__global__ __launch_bounds__(128) void flash_attn(const unsigned short* __restrict__ qb,
                                                  const unsigned short* __restrict__ kb,
                                                  const unsigned short* __restrict__ vb,
                                                  unsigned short* __restrict__ ob) {
    const int lane = threadIdx.x & 31;
    const int wave = threadIdx.x >> 5;
    const int qt   = blockIdx.x * 4 + wave;   // 16-row query tile index
    const int h    = blockIdx.y;
    const int b    = blockIdx.z;

    const size_t base = (size_t)b * SEQ * D_MODEL + (size_t)h * HD;
    const int half = lane >> 4;
    const int akb  = half ? 8 : 0;
    const int bkb  = half ? 8 : 0;

    // load Q fragments once (K = 64 contraction -> two 16x32 A fragments)
    const unsigned short* qrow = qb + base + (size_t)(qt * 16 + (lane & 15)) * D_MODEL;
    Frag aq0, aq1;
#pragma unroll
    for (int j = 0; j < 4; ++j) {
        aq0.u[j]     = *(const unsigned int*)(qrow + akb + 2 * j);
        aq0.u[j + 4] = *(const unsigned int*)(qrow + akb + 16 + 2 * j);
        aq1.u[j]     = *(const unsigned int*)(qrow + 32 + akb + 2 * j);
        aq1.u[j + 4] = *(const unsigned int*)(qrow + 32 + akb + 16 + 2 * j);
    }

    v8f acc[4];
#pragma unroll
    for (int t = 0; t < 4; ++t) acc[t] = (v8f){0.f, 0.f, 0.f, 0.f, 0.f, 0.f, 0.f, 0.f};
    float ms[8], ls[8];
#pragma unroll
    for (int j = 0; j < 8; ++j) { ms[j] = -1e30f; ls[j] = 0.f; }

    __shared__ unsigned short plds[4][16 * 32];
    unsigned short* pl = plds[wave];

    for (int c0 = 0; c0 < SEQ; c0 += 32) {
        // ---- scores: two 16x16 tiles over 32 keys
        v8f s[2];
#pragma unroll
        for (int t = 0; t < 2; ++t) {
            const unsigned short* krow =
                kb + base + (size_t)(c0 + t * 16 + (lane & 15)) * D_MODEL;
            Frag bk0, bk1;
#pragma unroll
            for (int j = 0; j < 8; ++j) {
                bk0.u[j] = *(const unsigned int*)(krow + bkb + 2 * j + ((j >> 2) << 3));
                bk1.u[j] = *(const unsigned int*)(krow + 32 + bkb + 2 * j + ((j >> 2) << 3));
            }
            v8f z = {0.f, 0.f, 0.f, 0.f, 0.f, 0.f, 0.f, 0.f};
            z    = __builtin_amdgcn_wmma_f32_16x16x32_bf16(false, aq0.v, false, bk0.v,
                                                           (short)0, z, false, false);
            s[t] = __builtin_amdgcn_wmma_f32_16x16x32_bf16(false, aq1.v, false, bk1.v,
                                                           (short)0, z, false, false);
#pragma unroll
            for (int j = 0; j < 8; ++j) s[t][j] *= 0.125f;  // 1/sqrt(64)
        }

        // ---- online softmax (row reductions across 16 lanes of this half)
        float rmax[8];
#pragma unroll
        for (int j = 0; j < 8; ++j) rmax[j] = fmaxf(s[0][j], s[1][j]);
#pragma unroll
        for (int m = 1; m <= 8; m <<= 1)
#pragma unroll
            for (int j = 0; j < 8; ++j) rmax[j] = fmaxf(rmax[j], __shfl_xor(rmax[j], m, 32));

        float alpha[8], rsum[8];
#pragma unroll
        for (int j = 0; j < 8; ++j) {
            float nm = fmaxf(ms[j], rmax[j]);
            alpha[j] = exp2f((ms[j] - nm) * 1.44269504f);
            ms[j]    = nm;
            rsum[j]  = 0.f;
        }
        v8f p[2];
#pragma unroll
        for (int t = 0; t < 2; ++t)
#pragma unroll
            for (int j = 0; j < 8; ++j) {
                p[t][j] = exp2f((s[t][j] - ms[j]) * 1.44269504f);
                rsum[j] += p[t][j];
            }
#pragma unroll
        for (int m = 1; m <= 8; m <<= 1)
#pragma unroll
            for (int j = 0; j < 8; ++j) rsum[j] += __shfl_xor(rsum[j], m, 32);
#pragma unroll
        for (int j = 0; j < 8; ++j) ls[j] = ls[j] * alpha[j] + rsum[j];
#pragma unroll
        for (int t = 0; t < 4; ++t)
#pragma unroll
            for (int j = 0; j < 8; ++j) acc[t][j] *= alpha[j];

        // ---- re-fragment P via LDS (C-layout -> A-layout)
#pragma unroll
        for (int t = 0; t < 2; ++t)
#pragma unroll
            for (int j = 0; j < 8; ++j)
                pl[(j + 8 * half) * 32 + t * 16 + (lane & 15)] = f2bf(p[t][j]);
        asm volatile("s_wait_dscnt 0" ::: "memory");

        Frag ap;
        {
            const unsigned short* prow = pl + (lane & 15) * 32;
#pragma unroll
            for (int j = 0; j < 4; ++j) {
                ap.u[j]     = *(const unsigned int*)(prow + akb + 2 * j);
                ap.u[j + 4] = *(const unsigned int*)(prow + akb + 16 + 2 * j);
            }
        }

        // ---- O += P @ V  (4 output subtiles of 16 cols)
#pragma unroll
        for (int t = 0; t < 4; ++t) {
            Frag bv;
            const int d = t * 16 + (lane & 15);
#pragma unroll
            for (int w = 0; w < 8; ++w) {
                const int k1 = c0 + bkb + 2 * w + ((w >> 2) << 3);
                bv.s[2 * w]     = vb[base + (size_t)k1 * D_MODEL + d];
                bv.s[2 * w + 1] = vb[base + (size_t)(k1 + 1) * D_MODEL + d];
            }
            acc[t] = __builtin_amdgcn_wmma_f32_16x16x32_bf16(false, ap.v, false, bv.v,
                                                             (short)0, acc[t], false, false);
        }
    }

    // ---- epilogue: normalize and store bf16
#pragma unroll
    for (int t = 0; t < 4; ++t)
#pragma unroll
        for (int j = 0; j < 8; ++j) {
            float o   = acc[t][j] / ls[j];
            int   row = qt * 16 + j + 8 * half;
            int   d   = t * 16 + (lane & 15);
            ob[base + (size_t)row * D_MODEL + d] = f2bf(o);
        }
}

// ---------------------------------------------------------------- layernorm (+residual)
__global__ __launch_bounds__(256) void ln_kernel(const float* __restrict__ xa,
                                                 const float* __restrict__ xb,
                                                 const float* __restrict__ g,
                                                 const float* __restrict__ bt,
                                                 float* __restrict__ yf,
                                                 unsigned short* __restrict__ yb) {
    const int t   = blockIdx.x;
    const int tid = threadIdx.x;
    __shared__ float s1[256], s2[256];
    float vals[4], lsum = 0.f, lsq = 0.f;
#pragma unroll
    for (int i = 0; i < 4; ++i) {
        int c = tid + i * 256;
        float v = xa[(size_t)t * D_MODEL + c] + xb[(size_t)t * D_MODEL + c];
        vals[i] = v; lsum += v; lsq += v * v;
    }
    s1[tid] = lsum; s2[tid] = lsq;
    __syncthreads();
    for (int off = 128; off > 0; off >>= 1) {
        if (tid < off) { s1[tid] += s1[tid + off]; s2[tid] += s2[tid + off]; }
        __syncthreads();
    }
    float mean = s1[0] * (1.f / D_MODEL);
    float var  = s2[0] * (1.f / D_MODEL) - mean * mean;
    float rstd = rsqrtf(var + 1e-5f);
#pragma unroll
    for (int i = 0; i < 4; ++i) {
        int c = tid + i * 256;
        float o = (vals[i] - mean) * rstd * g[c] + bt[c];
        if (yf) yf[(size_t)t * D_MODEL + c] = o;
        if (yb) yb[(size_t)t * D_MODEL + c] = f2bf(o);
    }
}

// ---------------------------------------------------------------- gate + categorical sample
__global__ __launch_bounds__(128) void gate_kernel(const float* __restrict__ x,
                                                   const float* __restrict__ gw,
                                                   const float* __restrict__ gb,
                                                   int* __restrict__ chosen) {
    const int lane = threadIdx.x & 31;
    const int t    = blockIdx.x * 4 + (threadIdx.x >> 5);
    float a[NEXP] = {0.f, 0.f, 0.f, 0.f};
    for (int d = lane; d < D_MODEL; d += 32) {
        float xv = x[(size_t)t * D_MODEL + d];
#pragma unroll
        for (int e = 0; e < NEXP; ++e) a[e] += xv * gw[e * D_MODEL + d];
    }
#pragma unroll
    for (int m = 1; m <= 16; m <<= 1)
#pragma unroll
        for (int e = 0; e < NEXP; ++e) a[e] += __shfl_xor(a[e], m, 32);
    if (lane == 0) {
        float mx = -1e30f;
#pragma unroll
        for (int e = 0; e < NEXP; ++e) { a[e] += gb[e]; mx = fmaxf(mx, a[e]); }
        float sum = 0.f, pr[NEXP];
#pragma unroll
        for (int e = 0; e < NEXP; ++e) { pr[e] = exp2f((a[e] - mx) * 1.44269504f); sum += pr[e]; }
        unsigned int h = (unsigned int)t * 0x9E3779B9u + 0x85EBCA6Bu;
        h ^= h >> 16; h *= 0x7FEB352Du; h ^= h >> 15; h *= 0x846CA68Bu; h ^= h >> 16;
        float u = (h >> 8) * (1.f / 16777216.f) * sum;
        int e = 0; float c = pr[0];
        while (e < NEXP - 1 && u >= c) { ++e; c += pr[e]; }
        chosen[t] = e;
    }
}

// ---------------------------------------------------------------- launch
extern "C" void kernel_launch(void* const* d_in, const int* in_sizes, int n_in,
                              void* d_out, int out_size, void* d_ws, size_t ws_size,
                              hipStream_t stream) {
    const float* src        = (const float*)d_in[0];
    const float* pos        = (const float*)d_in[1];
    const float* in_proj_w  = (const float*)d_in[2];
    const float* in_proj_b  = (const float*)d_in[3];
    const float* out_proj_w = (const float*)d_in[4];
    const float* out_proj_b = (const float*)d_in[5];
    const float* gate_w     = (const float*)d_in[6];
    const float* gate_b     = (const float*)d_in[7];
    const float* w1         = (const float*)d_in[8];
    const float* b1         = (const float*)d_in[9];
    const float* w2         = (const float*)d_in[10];
    const float* b2         = (const float*)d_in[11];
    const float* ln1_g      = (const float*)d_in[12];
    const float* ln1_b      = (const float*)d_in[13];
    const float* ln2_g      = (const float*)d_in[14];
    const float* ln2_b      = (const float*)d_in[15];

    char* wsp = (char*)d_ws;
    auto carve = [&](size_t bytes) {
        char* p = wsp;
        wsp += (bytes + 255) & ~(size_t)255;
        return p;
    };
    const size_t ND = (size_t)NTOK * D_MODEL;
    unsigned short* qk_b   = (unsigned short*)carve(ND * 2);
    unsigned short* src_b  = (unsigned short*)carve(ND * 2);
    unsigned short* q_b    = (unsigned short*)carve(ND * 2);
    unsigned short* k_b    = (unsigned short*)carve(ND * 2);
    unsigned short* v_b    = (unsigned short*)carve(ND * 2);
    unsigned short* o_b    = (unsigned short*)carve(ND * 2);
    unsigned short* x_b    = (unsigned short*)carve(ND * 2);
    unsigned short* h_b    = (unsigned short*)carve((size_t)NTOK * DFF_E * 2);
    float*          proj_f = (float*)carve(ND * 4);
    float*          x_f    = (float*)carve(ND * 4);
    float*          moe_f  = (float*)carve(ND * 4);
    unsigned short* wqkv_b = (unsigned short*)carve((size_t)3 * D_MODEL * D_MODEL * 2);
    unsigned short* wout_b = (unsigned short*)carve((size_t)D_MODEL * D_MODEL * 2);
    unsigned short* w1_b   = (unsigned short*)carve((size_t)NEXP * DFF_E * D_MODEL * 2);
    unsigned short* w2_b   = (unsigned short*)carve((size_t)NEXP * D_MODEL * DFF_E * 2);
    int*            chosen = (int*)carve((size_t)NTOK * 4);

    auto cvt_n = [&](const float* in, unsigned short* out, int n) {
        cvt_bf16<<<(n + 255) / 256, 256, 0, stream>>>(in, out, n);
    };
    // weight conversion
    cvt_n(in_proj_w, wqkv_b, 3 * D_MODEL * D_MODEL);
    cvt_n(out_proj_w, wout_b, D_MODEL * D_MODEL);
    cvt_n(w1, w1_b, NEXP * DFF_E * D_MODEL);
    cvt_n(w2, w2_b, NEXP * D_MODEL * DFF_E);

    // activations prep
    prep_kernel<<<(int)((ND + 255) / 256), 256, 0, stream>>>(src, pos, qk_b, src_b, (int)ND);

    // QKV projections (block tile 64x256)
    dim3 ggrid(D_MODEL / 256, NTOK / 64);
    gemm_bf16_nt<1><<<ggrid, 256, 0, stream>>>(qk_b, wqkv_b, in_proj_b, q_b,
                                               NTOK, D_MODEL, D_MODEL, nullptr, 0);
    gemm_bf16_nt<1><<<ggrid, 256, 0, stream>>>(qk_b, wqkv_b + (size_t)D_MODEL * D_MODEL,
                                               in_proj_b + D_MODEL, k_b,
                                               NTOK, D_MODEL, D_MODEL, nullptr, 0);
    gemm_bf16_nt<1><<<ggrid, 256, 0, stream>>>(src_b, wqkv_b + (size_t)2 * D_MODEL * D_MODEL,
                                               in_proj_b + 2 * D_MODEL, v_b,
                                               NTOK, D_MODEL, D_MODEL, nullptr, 0);

    // flash attention
    flash_attn<<<dim3(SEQ / 64, NHEAD, BATCH), 128, 0, stream>>>(q_b, k_b, v_b, o_b);

    // out projection -> f32, then LN1 (+residual src) -> x_f / x_b
    gemm_bf16_nt<0><<<ggrid, 256, 0, stream>>>(o_b, wout_b, out_proj_b, proj_f,
                                               NTOK, D_MODEL, D_MODEL, nullptr, 0);
    ln_kernel<<<NTOK, 256, 0, stream>>>(src, proj_f, ln1_g, ln1_b, x_f, x_b);

    // MoE gating
    gate_kernel<<<NTOK / 4, 128, 0, stream>>>(x_f, gate_w, gate_b, chosen);

    // MoE experts: h = relu(x@w1^T+b1) ; moe_out = h@w2^T+b2 for routed tokens
    for (int e = 0; e < NEXP; ++e) {
        gemm_bf16_nt<2><<<dim3(DFF_E / 256, NTOK / 64), 256, 0, stream>>>(
            x_b, w1_b + (size_t)e * DFF_E * D_MODEL, b1 + e * DFF_E, h_b,
            NTOK, DFF_E, D_MODEL, nullptr, 0);
        gemm_bf16_nt<3><<<dim3(D_MODEL / 256, NTOK / 64), 256, 0, stream>>>(
            h_b, w2_b + (size_t)e * D_MODEL * DFF_E, b2 + e * D_MODEL, moe_f,
            NTOK, D_MODEL, DFF_E, chosen, e);
    }

    // final LN -> output (f32)
    ln_kernel<<<NTOK, 256, 0, stream>>>(x_f, moe_f, ln2_g, ln2_b, (float*)d_out, nullptr);
}
